// CustomTransformer_9182640079597
// MI455X (gfx1250) — matrix operand order repaired and checked by reading, not compiled, not run on hardware
//
#include <hip/hip_runtime.h>
#include <hip/hip_bf16.h>

// ---------------------------------------------------------------------------
// CustomTransformer forward for MI455X (gfx1250).
// GEMMs: v_wmma_f32_16x16x32_bf16 (bf16 in, fp32 acc), LDS double-buffered
// via CDNA5 global_load_async_to_lds_b128 (ASYNCcnt pipeline).
// Softmax/gauss-bias/layernorm in fp32 VALU.
// ---------------------------------------------------------------------------

typedef __attribute__((ext_vector_type(16))) __bf16 v16bf;
typedef __attribute__((ext_vector_type(8)))  float  v8f;

#define TBM 64
#define TBN 128
#define TBK 32
#define LDSA (TBK + 8)   // padded LDS row stride (bf16 elems) -> 80B rows
#define LDSB (TBK + 8)

// One 16B chunk: async DMA global -> LDS, tracked by ASYNCcnt.
__device__ __forceinline__ void async_cp16(const __bf16* gptr, const __bf16* lptr)
{
  const unsigned lds  = (unsigned)(unsigned long long)lptr;      // addr[31:0] = LDS offset
  const unsigned long long ga = (unsigned long long)gptr;
  asm volatile("global_load_async_to_lds_b128 %0, %1, off"
               :: "v"(lds), "v"(ga) : "memory");
}

// C[M,N] = act(scale * (A @ Bt^T) + bias)
// A  : [M,K] bf16, row stride lda ; Bt : [N,K] bf16, row stride ldb.
// Batched over gridDim.z: z -> (bz=z/Hn, hz=z%Hn); ptr += bz*aB + hz*aH etc.
// Requires M%64==0, N%128==0, K%32==0 (true for every GEMM in this model).
__global__ __launch_bounds__(128) void gemm_nt_bf16(
    const __bf16* __restrict__ A, const __bf16* __restrict__ Bt,
    const float* __restrict__ bias,
    float* __restrict__ Cf, __bf16* __restrict__ Cb,
    int M, int N, int K, int lda, int ldb, int ldc,
    long long aB, long long aH, long long bB, long long bH,
    long long cB, long long cH, int Hn, float scale, int relu)
{
  (void)M; (void)N;
  const int z  = blockIdx.z;
  const int bz = z / Hn;
  const int hz = z % Hn;
  A  += (size_t)bz * aB + (size_t)hz * aH;
  Bt += (size_t)bz * bB + (size_t)hz * bH;
  const size_t coff = (size_t)bz * cB + (size_t)hz * cH;

  __shared__ __bf16 As[2][TBM * LDSA];   // 2 x 64x32 tile
  __shared__ __bf16 Bs[2][TBN * LDSB];   // 2 x 128x32 tile

  const int tid  = threadIdx.x;
  const int lane = tid & 31;
  const int wave = tid >> 5;
  const int bm = blockIdx.y * TBM;
  const int bn = blockIdx.x * TBN;
  const int wm = (wave >> 1) * 32;     // 2 waves along M
  const int wn = (wave & 1) * 64;      // 2 waves along N
  const int l16 = lane & 15;
  const int lhi = lane >> 4;

  v8f acc[2][4];
#pragma unroll
  for (int i = 0; i < 2; ++i)
#pragma unroll
    for (int j = 0; j < 4; ++j) acc[i][j] = (v8f){};

  // Issue one K-tile's async copies: A = 2 chunks/thread, B = 4 chunks/thread.
  auto issue = [&](int k0, int buf) {
#pragma unroll
    for (int u = 0; u < 2; ++u) {
      const int idx = tid + u * 128;           // 256 chunks
      const int r  = idx >> 2;
      const int cg = (idx & 3) * 8;
      async_cp16(A + (size_t)(bm + r) * lda + (k0 + cg),
                 &As[buf][r * LDSA + cg]);
    }
#pragma unroll
    for (int u = 0; u < 4; ++u) {
      const int idx = tid + u * 128;           // 512 chunks
      const int r  = idx >> 2;
      const int cg = (idx & 3) * 8;
      async_cp16(Bt + (size_t)(bn + r) * ldb + (k0 + cg),
                 &Bs[buf][r * LDSB + cg]);
    }
  };

  union Frag { v16bf v; float4 f[2]; };

  const int nk = K / TBK;
  issue(0, 0);

  for (int it = 0; it < nk; ++it) {
    const int buf = it & 1;
    if (it + 1 < nk) {
      issue((it + 1) * TBK, buf ^ 1);
      // 6 copies (next tile) still outstanding; previous tile complete.
      asm volatile("s_wait_asynccnt 0x6" ::: "memory");
    } else {
      asm volatile("s_wait_asynccnt 0x0" ::: "memory");
    }
    __syncthreads();

    if (it + 2 < nk) {   // warm L2 for tile it+2 (global_prefetch_b8)
      const int k2 = (it + 2) * TBK;
      __builtin_prefetch(A  + (size_t)(bm + (tid >> 1)) * lda + k2, 0, 3);
      __builtin_prefetch(Bt + (size_t)(bn + tid) * ldb + k2, 0, 3);
    }

    // A fragment (16x32): lane<16 -> K 0..7 & 16..23 ; lane>=16 -> K 8..15 & 24..31
    Frag a[2], b[4];
#pragma unroll
    for (int i = 0; i < 2; ++i) {
      const __bf16* p = &As[buf][(wm + i * 16 + l16) * LDSA + lhi * 8];
      a[i].f[0] = *(const float4*)p;
      a[i].f[1] = *(const float4*)(p + 16);
    }
    // B fragment (32x16): lane<16 -> K 0..7 & 16..23 ; lane>=16 -> K 8..15 & 24..31
#pragma unroll
    for (int j = 0; j < 4; ++j) {
      const __bf16* p = &Bs[buf][(wn + j * 16 + l16) * LDSB + lhi * 8];
      b[j].f[0] = *(const float4*)p;
      b[j].f[1] = *(const float4*)(p + 16);
    }

#pragma unroll
    for (int i = 0; i < 2; ++i)
#pragma unroll
      for (int j = 0; j < 4; ++j)
        acc[i][j] = __builtin_amdgcn_wmma_f32_16x16x32_bf16(
            false, a[i].v, false, b[j].v, (short)0, acc[i][j], false, false);

    __syncthreads();   // protect buf (just read) before next iteration's issue overwrites it
  }

  // Epilogue. C layout: VGPR r -> row r (lanes 0-15) / r+8 (lanes 16-31); col = lane&15.
#pragma unroll
  for (int i = 0; i < 2; ++i) {
    const int rb = bm + wm + i * 16 + lhi * 8;
#pragma unroll
    for (int j = 0; j < 4; ++j) {
      const int col = bn + wn + j * 16 + l16;
      const float bia = bias ? bias[col] : 0.0f;
#pragma unroll
      for (int r = 0; r < 8; ++r) {
        float v = acc[i][j][r] * scale + bia;
        if (relu) v = fmaxf(v, 0.0f);
        const size_t o = coff + (size_t)(rb + r) * ldc + col;
        if (Cf) Cf[o] = v;
        if (Cb) Cb[o] = (__bf16)v;
      }
    }
  }
}

// dst[C,R] (bf16) = transpose(src[R,C] (f32)), batched over z.
__global__ void transpose_cvt_kernel(const float* __restrict__ src,
                                     __bf16* __restrict__ dst,
                                     int R, int C, long long sB, long long dB)
{
  src += (size_t)blockIdx.z * sB;
  dst += (size_t)blockIdx.z * dB;
  const int c = blockIdx.x * 32 + threadIdx.x;
  const int r = blockIdx.y * 8 + threadIdx.y;
  if (r < R && c < C)
    dst[(size_t)c * R + r] = (__bf16)src[(size_t)r * C + c];
}

__global__ __launch_bounds__(256) void cvt_bf16_kernel(const float* __restrict__ s,
                                                       __bf16* __restrict__ d, size_t n)
{
  const size_t i = (size_t)blockIdx.x * 256 + threadIdx.x;
  if (i < n) d[i] = (__bf16)s[i];
}

// Fused softmax * gaussian positional bias * renormalize, per score row.
// w_j = exp(s_j - max - (q-j)^2 * inv2sig2) / sum.  Tk fixed at 512.
__global__ __launch_bounds__(256) void softmax_gauss_kernel(
    const float* __restrict__ S, __bf16* __restrict__ W, int Tq, float inv2sig2)
{
  const int Tk = 512;
  const int q = blockIdx.x;
  const size_t row = ((size_t)blockIdx.y * Tq + q) * (size_t)Tk;
  const int t = threadIdx.x;
  __shared__ float red[256];

  const float v0 = S[row + t];
  const float v1 = S[row + t + 256];
  red[t] = fmaxf(v0, v1);
  __syncthreads();
  for (int st = 128; st > 0; st >>= 1) {
    if (t < st) red[t] = fmaxf(red[t], red[t + st]);
    __syncthreads();
  }
  const float m = red[0];
  __syncthreads();

  const float d0 = (float)(q - t);
  const float d1 = (float)(q - (t + 256));
  const float e0 = __expf(v0 - m - d0 * d0 * inv2sig2);
  const float e1 = __expf(v1 - m - d1 * d1 * inv2sig2);
  red[t] = e0 + e1;
  __syncthreads();
  for (int st = 128; st > 0; st >>= 1) {
    if (t < st) red[t] += red[t + st];
    __syncthreads();
  }
  const float inv = 1.0f / red[0];
  W[row + t]       = (__bf16)(e0 * inv);
  W[row + t + 256] = (__bf16)(e1 * inv);
}

// out = LayerNorm(x + res) * g + b, emits fp32 (residual stream) + bf16.  E=512.
__global__ __launch_bounds__(256) void add_ln_kernel(
    const float* __restrict__ x, const float* __restrict__ res,
    const float* __restrict__ g, const float* __restrict__ b,
    float* __restrict__ of, __bf16* __restrict__ ob)
{
  const size_t base = (size_t)blockIdx.x * 512;
  const int t = threadIdx.x;
  __shared__ float red[256];

  const float a0 = x[base + t] + res[base + t];
  const float a1 = x[base + t + 256] + res[base + t + 256];
  red[t] = a0 + a1;
  __syncthreads();
  for (int st = 128; st > 0; st >>= 1) {
    if (t < st) red[t] += red[t + st];
    __syncthreads();
  }
  const float mu = red[0] * (1.0f / 512.0f);
  __syncthreads();

  const float c0 = a0 - mu, c1 = a1 - mu;
  red[t] = c0 * c0 + c1 * c1;
  __syncthreads();
  for (int st = 128; st > 0; st >>= 1) {
    if (t < st) red[t] += red[t + st];
    __syncthreads();
  }
  const float rstd = rsqrtf(red[0] * (1.0f / 512.0f) + 1e-5f);

  const float o0 = c0 * rstd * g[t] + b[t];
  const float o1 = c1 * rstd * g[t + 256] + b[t + 256];
  of[base + t] = o0;
  of[base + t + 256] = o1;
  if (ob) { ob[base + t] = (__bf16)o0; ob[base + t + 256] = (__bf16)o1; }
}

// ---------------------------------------------------------------------------

extern "C" void kernel_launch(void* const* d_in, const int* in_sizes, int n_in,
                              void* d_out, int out_size, void* d_ws, size_t ws_size,
                              hipStream_t stream)
{
  (void)in_sizes; (void)n_in; (void)out_size; (void)ws_size;

  const int Bq = 8, T = 512, E = 512, Hh = 8, hd = 64, F = 2048;
  const int Mtok = Bq * T;                       // 4096 token rows
  const float inv2sig2 = 1.0f / (2.0f * (T / 4.0f) * (T / 4.0f));

  const float* src = (const float*)d_in[0];
  const float* tgt = (const float*)d_in[1];

  // ---- parse flattened params (setup_inputs insertion order) ----
  struct AttnP { const float *Wq,*bq,*Wk,*bk,*lw,*lb,*ow,*obp; };
  struct LnP   { const float *g,*b; };
  struct FfnP  { const float *w1,*b1,*w2,*b2; };
  int cur = 2;
  auto nxt = [&]() -> const float* { return (const float*)d_in[cur++]; };
  auto rdA = [&]() { AttnP a; a.Wq=nxt(); a.bq=nxt(); a.Wk=nxt(); a.bk=nxt();
                     a.lw=nxt(); a.lb=nxt(); a.ow=nxt(); a.obp=nxt(); return a; };
  auto rdL = [&]() { LnP p; p.g=nxt(); p.b=nxt(); return p; };
  auto rdF = [&]() { FfnP f; f.w1=nxt(); f.b1=nxt(); f.w2=nxt(); f.b2=nxt(); return f; };

  AttnP esa[3]; LnP eln1[3], eln2[3]; FfnP eff[3];
  AttnP dsa[3], dca[3]; LnP dln1[3], dln2[3], dln3[3]; FfnP dff[3];
  for (int l = 0; l < 3; ++l) { esa[l]=rdA(); eln1[l]=rdL(); eln2[l]=rdL(); eff[l]=rdF(); }
  for (int l = 0; l < 3; ++l) { dsa[l]=rdA(); dca[l]=rdA(); dln1[l]=rdL();
                                dln2[l]=rdL(); dln3[l]=rdL(); dff[l]=rdF(); }

  // ---- workspace bump allocator ----
  char* wp = (char*)d_ws;
  auto alloc = [&](size_t bytes) -> void* {
    void* p = wp; wp += (bytes + 255) & ~(size_t)255; return p;
  };
  float*  xf  = (float*) alloc((size_t)Mtok * E * 4);          // encoder residual (fp32)
  __bf16* xb  = (__bf16*)alloc((size_t)Mtok * E * 2);
  float*  yf  = (float*) alloc((size_t)Mtok * E * 4);          // decoder residual
  __bf16* yb  = (__bf16*)alloc((size_t)Mtok * E * 2);
  __bf16* qb  = (__bf16*)alloc((size_t)Mtok * E * 2);
  __bf16* kb  = (__bf16*)alloc((size_t)Mtok * E * 2);
  __bf16* vT  = (__bf16*)alloc((size_t)Bq * E * T * 2);        // [B,E,T]
  float*  sc  = (float*) alloc((size_t)Bq * Hh * T * T * 4);   // scores
  __bf16* wt  = (__bf16*)alloc((size_t)Bq * Hh * T * T * 2);   // attn weights
  __bf16* ao  = (__bf16*)alloc((size_t)Mtok * Hh * E * 2);     // [B,Tq,H*E] concat
  __bf16* t1  = (__bf16*)alloc((size_t)Mtok * E * 2);          // after lin
  float*  af  = (float*) alloc((size_t)Mtok * E * 4);          // block output fp32
  __bf16* wT  = (__bf16*)alloc((size_t)Hh * E * E * 2);        // weight^T scratch (4MB)
  __bf16* h1  = (__bf16*)alloc((size_t)Mtok * F * 2);          // FFN hidden

  // ---- launch helpers ----
  auto cvt = [&](const float* s, __bf16* d, size_t n) {
    cvt_bf16_kernel<<<dim3((unsigned)((n + 255) / 256)), dim3(256), 0, stream>>>(s, d, n);
  };
  auto tr = [&](const float* s, __bf16* d, int R, int C, long long sB, long long dB, int Z) {
    transpose_cvt_kernel<<<dim3((C + 31) / 32, (R + 7) / 8, Z), dim3(32, 8), 0, stream>>>(
        s, d, R, C, sB, dB);
  };
  auto gemm = [&](const __bf16* A, const __bf16* Bt, const float* bias,
                  float* Cf, __bf16* Cb, int M, int N, int K,
                  int lda, int ldb, int ldc,
                  long long aB, long long aH, long long bB, long long bH,
                  long long cB, long long cH, int Hn, int Z, float scale, int relu) {
    gemm_nt_bf16<<<dim3(N / TBN, M / TBM, Z), dim3(128), 0, stream>>>(
        A, Bt, bias, Cf, Cb, M, N, K, lda, ldb, ldc, aB, aH, bB, bH, cB, cH,
        Hn, scale, relu);
  };
  auto addln = [&](const float* xin, const float* res, const LnP& p,
                   float* of, __bf16* ob) {
    add_ln_kernel<<<dim3(Mtok), dim3(256), 0, stream>>>(xin, res, p.g, p.b, of, ob);
  };

  // gauss_attn(q_in, k_in, v_in): q/k projections, QK^T, softmax*gauss,
  // wts @ raw-v broadcast over heads, lin, op.
  auto attention = [&](const AttnP& p, const __bf16* q_b,
                       const float* kv_f, const __bf16* kv_b, float* outf) {
    tr(p.Wq, wT, E, E, 0, 0, 1);
    gemm(q_b, wT, p.bq, nullptr, qb, Mtok, E, E, E, E, E,
         0, 0, 0, 0, 0, 0, 1, 1, 1.0f, 0);
    tr(p.Wk, wT, E, E, 0, 0, 1);
    gemm(kv_b, wT, p.bk, nullptr, kb, Mtok, E, E, E, E, E,
         0, 0, 0, 0, 0, 0, 1, 1, 1.0f, 0);
    // vT[b] = kv[b]^T  (raw, unprojected values)
    tr(kv_f, vT, T, E, (long long)T * E, (long long)E * T, Bq);
    // scores[b,h] = q[b,:,h*64:] @ k[b,:,h*64:]^T / 8
    gemm(qb, kb, nullptr, sc, nullptr, T, T, hd, E, E, T,
         (long long)T * E, hd, (long long)T * E, hd,
         (long long)Hh * T * T, (long long)T * T, Hh, Bq * Hh, 0.125f, 0);
    softmax_gauss_kernel<<<dim3(T, Bq * Hh), dim3(256), 0, stream>>>(sc, wt, T, inv2sig2);
    // ao[b,q,h*E+e] = wts[b,h] @ v[b]
    gemm(wt, vT, nullptr, nullptr, ao, T, E, T, T, T, Hh * E,
         (long long)Hh * T * T, (long long)T * T, (long long)E * T, 0,
         (long long)T * Hh * E, (long long)E, Hh, Bq * Hh, 1.0f, 0);
    // lin: [Mtok, H*E] @ [H*E, E]
    tr(p.lw, wT, Hh * E, E, 0, 0, 1);
    gemm(ao, wT, p.lb, nullptr, t1, Mtok, E, Hh * E, Hh * E, Hh * E, E,
         0, 0, 0, 0, 0, 0, 1, 1, 1.0f, 0);
    // op: [Mtok, E] @ [E, E]
    tr(p.ow, wT, E, E, 0, 0, 1);
    gemm(t1, wT, p.obp, outf, nullptr, Mtok, E, E, E, E, E,
         0, 0, 0, 0, 0, 0, 1, 1, 1.0f, 0);
  };

  auto ffn = [&](const FfnP& p, const __bf16* xin_b, float* outf) {
    tr(p.w1, wT, E, F, 0, 0, 1);
    gemm(xin_b, wT, p.b1, nullptr, h1, Mtok, F, E, E, E, F,
         0, 0, 0, 0, 0, 0, 1, 1, 1.0f, 1 /*relu*/);
    tr(p.w2, wT, F, E, 0, 0, 1);
    gemm(h1, wT, p.b2, outf, nullptr, Mtok, E, F, F, F, E,
         0, 0, 0, 0, 0, 0, 1, 1, 1.0f, 0);
  };

  // ---- encoder ----
  hipMemcpyAsync(xf, src, (size_t)Mtok * E * 4, hipMemcpyDeviceToDevice, stream);
  cvt(src, xb, (size_t)Mtok * E);
  for (int l = 0; l < 3; ++l) {
    attention(esa[l], xb, xf, xb, af);
    addln(af, xf, eln1[l], xf, xb);
    ffn(eff[l], xb, af);
    addln(af, xf, eln2[l], xf, xb);
  }
  // xf/xb now hold encoder memory.

  // ---- decoder ----
  hipMemcpyAsync(yf, tgt, (size_t)Mtok * E * 4, hipMemcpyDeviceToDevice, stream);
  cvt(tgt, yb, (size_t)Mtok * E);
  for (int l = 0; l < 3; ++l) {
    attention(dsa[l], yb, yf, yb, af);            // self-attn
    addln(af, yf, dln1[l], yf, yb);
    attention(dca[l], yb, xf, xb, af);            // cross-attn (kv = memory)
    addln(af, yf, dln2[l], yf, yb);
    ffn(dff[l], yb, af);
    float* dst = (l == 2) ? (float*)d_out : yf;   // final LN writes result
    addln(af, yf, dln3[l], dst, yb);
  }
}
